// TextTimeCrystal_75814762709389
// MI455X (gfx1250) — matrix operand order repaired and checked by reading, not compiled, not run on hardware
//
#include <hip/hip_runtime.h>
#include <math.h>

// ---------------- problem constants ----------------
#define BB   32
#define SS   512
#define DD   1024
#define VV   50000
#define MM   (BB*SS)          // 16384 rows
#define TMEM 64

typedef __attribute__((ext_vector_type(16))) __bf16        v16bf;
typedef __attribute__((ext_vector_type(8)))  float         v8f;
typedef __attribute__((ext_vector_type(8)))  unsigned int  v8u;

// ---------------- helpers ----------------
__device__ __forceinline__ unsigned short f2bf(float f) {
    unsigned int u = __float_as_uint(f);
    unsigned int r = u + 0x7FFFu + ((u >> 16) & 1u);   // RNE
    return (unsigned short)(r >> 16);
}
__device__ __forceinline__ float bf2f(unsigned short h) {
    return __uint_as_float(((unsigned int)h) << 16);
}
__device__ __forceinline__ unsigned int pack2(unsigned short a, unsigned short b) {
    return (unsigned int)a | ((unsigned int)b << 16);
}
__device__ __forceinline__ float dev_osc(int ts) {  // ts = time_step + 1
    const double TWO_PI = 6.283185307179586476925287;
    double t  = (double)ts / 100.0;
    double pp = fmod(TWO_PI * t * 1.618033, TWO_PI);
    double sp = fmod(pp / 1.618033, TWO_PI);
    return (float)(0.3 * (sin(pp) + 0.3 * cos(sp)));
}
// full-block (1024 thr / 32 waves) sum reduction; result valid in thread 0
__device__ __forceinline__ float blockReduceSum(float v, float* s_part) {
    #pragma unroll
    for (int off = 16; off > 0; off >>= 1) v += __shfl_down(v, off, 32);
    int lane = threadIdx.x & 31, w = threadIdx.x >> 5;
    __syncthreads();
    if (lane == 0) s_part[w] = v;
    __syncthreads();
    float r = 0.f;
    if (threadIdx.x < 32) {
        r = s_part[threadIdx.x];
        #pragma unroll
        for (int off = 16; off > 0; off >>= 1) r += __shfl_down(r, off, 32);
    }
    return r;
}

// ---------------- small kernels ----------------
__global__ void ttc_zero_counts(unsigned int* __restrict__ counts) {
    int i = blockIdx.x * blockDim.x + threadIdx.x;
    if (i < VV) counts[i] = 0u;
}

__global__ void ttc_hist(const int* __restrict__ ids, unsigned int* __restrict__ counts) {
    int i = blockIdx.x * blockDim.x + threadIdx.x;
    if (i < MM) {
        int id = ids[i];
        if (id >= 0 && id < VV) atomicAdd(&counts[id], 1u);
    }
}

__global__ void ttc_vocab(const unsigned int* __restrict__ counts,
                          const float* __restrict__ vr,
                          const int* __restrict__ time_step,
                          float* __restrict__ out_vr) {
    int i = blockIdx.x * blockDim.x + threadIdx.x;
    if (i >= VV) return;
    float osc = dev_osc(time_step[0] + 1);
    float d = powf(0.9f, (float)counts[i]);
    out_vr[i] = d * vr[i] + (1.0f - d) * osc;
}

// two-phase deterministic mean over (B,S): 64 row-blocks of 256 rows each
__global__ __launch_bounds__(256)
void ttc_avg_partial(const float* __restrict__ x, float* __restrict__ partials) {
    const int rb = blockIdx.x;          // 0..63
    const int t  = threadIdx.x;         // 0..255
    const float* base = x + (size_t)rb * 256 * DD;
    float a0 = 0.f, a1 = 0.f, a2 = 0.f, a3 = 0.f;
    for (int r = 0; r < 256; ++r) {
        const float* row = base + (size_t)r * DD;
        a0 += row[t];       a1 += row[t + 256];
        a2 += row[t + 512]; a3 += row[t + 768];
    }
    float* p = partials + (size_t)rb * DD;
    p[t] = a0; p[t + 256] = a1; p[t + 512] = a2; p[t + 768] = a3;
}

// one block, 1024 threads: avg finalize, matvec+tanh, new_cs, new_sm,
// cosine-sim chain, coherence, gate vector for the big kernel
__global__ __launch_bounds__(1024)
void ttc_finalize(const float* __restrict__ partials,
                  const float* __restrict__ crystal_state,
                  const float* __restrict__ semantic_memory,
                  const float* __restrict__ coherence_buffer,
                  const float* __restrict__ crystal_weights,
                  const float* __restrict__ W_se,
                  const float* __restrict__ b_se,
                  const int*   __restrict__ time_step,
                  float* __restrict__ gate,
                  float* __restrict__ out_new_cs,
                  float* __restrict__ out_new_sm,
                  float* __restrict__ out_new_cb) {
    __shared__ float s_avg[DD];
    __shared__ float s_part[32];
    __shared__ float s_norm[10];
    __shared__ float s_dot[9];
    __shared__ float s_coh;

    const int t = threadIdx.x;

    float sum = 0.f;
    #pragma unroll 4
    for (int r = 0; r < 64; ++r) sum += partials[r * DD + t];
    float avg = sum * (1.0f / (float)MM);
    s_avg[t] = avg;
    out_new_sm[t] = avg;                                   // row 0 = avg
    for (int r = 0; r < TMEM - 1; ++r)                     // shift memory
        out_new_sm[(r + 1) * DD + t] = semantic_memory[r * DD + t];
    __syncthreads();

    // text_mod[t] = tanh(avg . W_se[t,:] + b_se[t])
    float accv = b_se[t];
    for (int k = 0; k < DD; ++k)
        accv = fmaf(s_avg[k], W_se[(size_t)t * DD + k], accv);
    float tm = tanhf(accv);

    int   ts  = time_step[0] + 1;
    float osc = dev_osc(ts);
    float ncs = 0.98f * crystal_state[t] + 0.02f * osc * crystal_weights[t] * (1.0f + tm);
    out_new_cs[t] = ncs;

    // cosine similarity over first min(ts,10) rows of new_sm
    int nrows = ts < 10 ? ts : 10;
    for (int r = 0; r < nrows; ++r) {
        float v = (r == 0) ? s_avg[t] : semantic_memory[(r - 1) * DD + t];
        float red = blockReduceSum(v * v, s_part);
        if (t == 0) s_norm[r] = sqrtf(red);
    }
    for (int p = 0; p + 1 < nrows; ++p) {
        float a = (p == 0) ? s_avg[t] : semantic_memory[(p - 1) * DD + t];
        float b = semantic_memory[p * DD + t];   // row p+1
        float red = blockReduceSum(a * b, s_part);
        if (t == 0) s_dot[p] = red;
    }
    __syncthreads();
    if (t == 0) {
        int np = nrows - 1;
        float sim = 0.f;
        for (int p = 0; p < np; ++p)
            sim += s_dot[p] / fmaxf(s_norm[p] * s_norm[p + 1], 1e-8f);
        if (np > 0) sim /= (float)np;
        float coh = 0.9f * (1.0f + 0.3f * sim);
        coh = fminf(fmaxf(coh, 0.5f), 1.5f);
        s_coh = coh;
        out_new_cb[0] = coh;
    }
    __syncthreads();
    if (t < TMEM - 1) out_new_cb[t + 1] = coherence_buffer[t];
    gate[t] = ncs * s_coh * 0.15f;                 // new_cs * coh * SEM_COUP
}

// ---------------- flagship: fused WMMA GEMM + sigmoid gate + residual ----------------
// out[m,n] = X[m,n] + sigmoid( sum_k X[m,k]*W[n,k] + bias[n] ) * gate[n]
// 128x128 block tile, K-tile 32, 8 waves: wave = 32(M) x 64(N), split-bf16 (3 WMMAs)
__global__ __launch_bounds__(256)
void ttc_gemm_gate(const float* __restrict__ X, const float* __restrict__ W,
                   const float* __restrict__ bias, const float* __restrict__ gate,
                   float* __restrict__ out) {
    __shared__ unsigned int sAhi[128 * 17];
    __shared__ unsigned int sAlo[128 * 17];
    __shared__ unsigned int sWhi[128 * 17];
    __shared__ unsigned int sWlo[128 * 17];
    __shared__ float sGs[128];
    __shared__ float sBias[128];

    const int tid = threadIdx.x;
    const int m0 = blockIdx.x * 128;
    const int n0 = blockIdx.y * 128;

    if (tid < 128) { sGs[tid] = gate[n0 + tid]; sBias[tid] = bias[n0 + tid]; }

    const int lane = tid & 31;
    const int wave = tid >> 5;
    const int wm   = (wave >> 1) * 32;     // 0,32,64,96
    const int wn   = (wave & 1) * 64;      // 0,64
    const int half = lane >> 4;
    const int l16  = lane & 15;

    v8f acc[2][4];
    #pragma unroll
    for (int mt = 0; mt < 2; ++mt)
        #pragma unroll
        for (int nt = 0; nt < 4; ++nt)
            #pragma unroll
            for (int i = 0; i < 8; ++i) acc[mt][nt][i] = 0.0f;

    const int rStage = tid >> 3;        // 0..31
    const int c4     = (tid & 7) * 4;   // float offset within K-tile

    for (int kt = 0; kt < DD; kt += 32) {
        // ---- stage fp32 -> bf16 hi/lo into LDS ----
        #pragma unroll
        for (int pass = 0; pass < 4; ++pass) {
            int r = rStage + pass * 32;
            float4 xv = *(const float4*)(X + (size_t)(m0 + r) * DD + kt + c4);
            float4 wv = *(const float4*)(W + (size_t)(n0 + r) * DD + kt + c4);

            unsigned short xh0 = f2bf(xv.x), xh1 = f2bf(xv.y), xh2 = f2bf(xv.z), xh3 = f2bf(xv.w);
            sAhi[r * 17 + (c4 >> 1) + 0] = pack2(xh0, xh1);
            sAhi[r * 17 + (c4 >> 1) + 1] = pack2(xh2, xh3);
            sAlo[r * 17 + (c4 >> 1) + 0] = pack2(f2bf(xv.x - bf2f(xh0)), f2bf(xv.y - bf2f(xh1)));
            sAlo[r * 17 + (c4 >> 1) + 1] = pack2(f2bf(xv.z - bf2f(xh2)), f2bf(xv.w - bf2f(xh3)));

            unsigned short wh0 = f2bf(wv.x), wh1 = f2bf(wv.y), wh2 = f2bf(wv.z), wh3 = f2bf(wv.w);
            sWhi[r * 17 + (c4 >> 1) + 0] = pack2(wh0, wh1);
            sWhi[r * 17 + (c4 >> 1) + 1] = pack2(wh2, wh3);
            sWlo[r * 17 + (c4 >> 1) + 0] = pack2(f2bf(wv.x - bf2f(wh0)), f2bf(wv.y - bf2f(wh1)));
            sWlo[r * 17 + (c4 >> 1) + 1] = pack2(f2bf(wv.z - bf2f(wh2)), f2bf(wv.w - bf2f(wh3)));
        }
        if (kt + 32 < DD) {   // hint next K tile toward cache (global_prefetch_b8)
            __builtin_prefetch((const void*)(X + (size_t)(m0 + rStage) * DD + kt + 32 + c4), 0, 1);
            __builtin_prefetch((const void*)(W + (size_t)(n0 + rStage) * DD + kt + 32 + c4), 0, 1);
        }
        __syncthreads();

        // ---- build fragments from LDS (CDNA5 wave32 layouts) ----
        v8u ahi[2], alo[2], bhi[4], blo[4];
        #pragma unroll
        for (int mt = 0; mt < 2; ++mt) {
            int row = wm + mt * 16 + l16;
            #pragma unroll
            for (int v = 0; v < 8; ++v) {
                int ku = (v & 3) + half * 4 + (v >> 2) * 8;   // A 16x32 layout (dword units)
                ahi[mt][v] = sAhi[row * 17 + ku];
                alo[mt][v] = sAlo[row * 17 + ku];
            }
        }
        #pragma unroll
        for (int nt = 0; nt < 4; ++nt) {
            int row = wn + nt * 16 + l16;
            #pragma unroll
            for (int v = 0; v < 8; ++v) {
                int ku = (v & 3) + half * 4 + (v >> 2) * 8;   // B 32x16 layout (dword units)
                bhi[nt][v] = sWhi[row * 17 + ku];
                blo[nt][v] = sWlo[row * 17 + ku];
            }
        }

        // ---- split-bf16 WMMA: hi*hi + hi*lo + lo*hi, fp32 accumulate ----
        #pragma unroll
        for (int mt = 0; mt < 2; ++mt) {
            v16bf A  = __builtin_bit_cast(v16bf, ahi[mt]);
            v16bf Al = __builtin_bit_cast(v16bf, alo[mt]);
            #pragma unroll
            for (int nt = 0; nt < 4; ++nt) {
                v16bf Bh = __builtin_bit_cast(v16bf, bhi[nt]);
                v16bf Bl = __builtin_bit_cast(v16bf, blo[nt]);
                acc[mt][nt] = __builtin_amdgcn_wmma_f32_16x16x32_bf16(
                    false, A,  false, Bh, (short)0, acc[mt][nt], false, false);
                acc[mt][nt] = __builtin_amdgcn_wmma_f32_16x16x32_bf16(
                    false, A,  false, Bl, (short)0, acc[mt][nt], false, false);
                acc[mt][nt] = __builtin_amdgcn_wmma_f32_16x16x32_bf16(
                    false, Al, false, Bh, (short)0, acc[mt][nt], false, false);
            }
        }
        __syncthreads();
    }

    // ---- fused epilogue: sigmoid gate + residual add ----
    #pragma unroll
    for (int mt = 0; mt < 2; ++mt)
        #pragma unroll
        for (int nt = 0; nt < 4; ++nt)
            #pragma unroll
            for (int r = 0; r < 8; ++r) {
                int gm = m0 + wm + mt * 16 + half * 8 + r;   // C/D layout: VGPR r -> M
                int nl = wn + nt * 16 + l16;                 // lane -> N
                size_t idx = (size_t)gm * DD + (n0 + nl);
                float s  = acc[mt][nt][r] + sBias[nl];
                float cf = 1.0f / (1.0f + expf(-s));
                out[idx] = X[idx] + cf * sGs[nl];
            }
}

// ---------------- launcher ----------------
extern "C" void kernel_launch(void* const* d_in, const int* in_sizes, int n_in,
                              void* d_out, int out_size, void* d_ws, size_t ws_size,
                              hipStream_t stream) {
    const float* x                = (const float*)d_in[0];   // (B,S,D)
    const float* crystal_state    = (const float*)d_in[1];   // (D,)
    const float* semantic_memory  = (const float*)d_in[2];   // (64,D)
    const float* coherence_buffer = (const float*)d_in[3];   // (64,)
    const float* vocab_resonance  = (const float*)d_in[4];   // (V,)
    const float* crystal_weights  = (const float*)d_in[5];   // (D,)
    const float* W_se             = (const float*)d_in[6];   // (D,D)
    const float* b_se             = (const float*)d_in[7];   // (D,)
    const float* W_cg             = (const float*)d_in[8];   // (D,D)
    const float* b_cg             = (const float*)d_in[9];   // (D,)
    const int*   token_ids        = (const int*)d_in[10];    // (B,S)
    const int*   time_step        = (const int*)d_in[11];    // scalar

    float* out_enhanced = (float*)d_out;                       // B*S*D
    float* out_new_cs   = out_enhanced + (size_t)MM * DD;      // D
    float* out_new_sm   = out_new_cs + DD;                     // 64*D
    float* out_new_cb   = out_new_sm + (size_t)TMEM * DD;      // 64
    float* out_new_vr   = out_new_cb + TMEM;                   // V

    char* ws = (char*)d_ws;
    float*        partials = (float*)ws;                               // 64*1024 f32
    unsigned int* counts   = (unsigned int*)(ws + 64 * DD * 4);        // V u32 (pad 50048)
    float*        gatevec  = (float*)(ws + 64 * DD * 4 + 50048 * 4);   // D f32

    ttc_zero_counts<<<(VV + 255) / 256, 256, 0, stream>>>(counts);
    ttc_avg_partial<<<64, 256, 0, stream>>>(x, partials);
    ttc_hist<<<(MM + 255) / 256, 256, 0, stream>>>(token_ids, counts);
    ttc_finalize<<<1, 1024, 0, stream>>>(partials, crystal_state, semantic_memory,
                                         coherence_buffer, crystal_weights, W_se, b_se,
                                         time_step, gatevec, out_new_cs, out_new_sm,
                                         out_new_cb);
    ttc_vocab<<<(VV + 255) / 256, 256, 0, stream>>>(counts, vocab_resonance, time_step,
                                                    out_new_vr);
    ttc_gemm_gate<<<dim3(MM / 128, DD / 128), 256, 0, stream>>>(x, W_cg, b_cg, gatevec,
                                                                out_enhanced);
}
